// CausalSelfAttention_60662118088962
// MI455X (gfx1250) — compile-verified, matrix operand order checked by reading, not run on hardware
//
#include <hip/hip_runtime.h>
#include <hip/hip_bf16.h>

// ---------------------------------------------------------------------------
// CDNA5 / gfx1250 causal self-attention, bf16 WMMA pipeline.
//   B=4, T=2048, C=1024, H=16, HD=64
// ---------------------------------------------------------------------------

typedef __attribute__((ext_vector_type(16))) __bf16 v16bf;
typedef __attribute__((ext_vector_type(8)))  __bf16 v8bf;
typedef __attribute__((ext_vector_type(4)))  __bf16 v4bf;
typedef __attribute__((ext_vector_type(8)))  float  v8f;

#define WMMA_BF16(a, b, c) \
  __builtin_amdgcn_wmma_f32_16x16x32_bf16(false, (a), false, (b), (short)0, (c), false, false)

static __device__ __forceinline__ v16bf mkfrag(v8bf lo, v8bf hi) {
  v16bf f;
#pragma unroll
  for (int e = 0; e < 8; ++e) { f[e] = lo[e]; f[e + 8] = hi[e]; }
  return f;
}

static __device__ __forceinline__ v8f v8f_zero() {
  v8f z;
#pragma unroll
  for (int e = 0; e < 8; ++e) z[e] = 0.0f;
  return z;
}

// ---------------------------------------------------------------------------
// fp32 -> bf16 conversion (4 elements per thread, vectorized load/store)
// ---------------------------------------------------------------------------
__global__ __launch_bounds__(256) void cvt_f32_bf16_kernel(const float* __restrict__ src,
                                                           __bf16* __restrict__ dst) {
  int i = (blockIdx.x * 256 + threadIdx.x) * 4;
  float4 v = *reinterpret_cast<const float4*>(src + i);
  v4bf o;
  o[0] = (__bf16)v.x; o[1] = (__bf16)v.y; o[2] = (__bf16)v.z; o[3] = (__bf16)v.w;
  *reinterpret_cast<v4bf*>(dst + i) = o;
}

// ---------------------------------------------------------------------------
// Shared WMMA GEMM core:  C[M,N] = A[M,K] * B[N,K]^T   (both row-major, bf16)
// Block tile 64(M) x 128(N), K-step 64, double-buffered LDS (1 barrier/iter),
// 8 wave32s as 2x4, each wave 32x32 (8 WMMAs per K-step).
// LDS rows padded to 72 elements (144B, 16B-aligned) for clean ds_load_b128.
// ---------------------------------------------------------------------------
static __device__ __forceinline__ void gemm_core(const __bf16* __restrict__ A,
                                                 const __bf16* __restrict__ B,
                                                 int K, int m0, int n0,
                                                 v8f acc[2][2]) {
  __shared__ __bf16 As[2][64 * 72];
  __shared__ __bf16 Bs[2][128 * 72];

  const int tid  = threadIdx.x;
  const int lane = tid & 31;
  const int nloc = lane & 15;
  const int half = lane >> 4;
  const int widx = tid >> 5;
  const int wm   = widx >> 2;   // 0..1
  const int wn   = widx & 3;    // 0..3

#pragma unroll
  for (int i = 0; i < 2; ++i)
#pragma unroll
    for (int j = 0; j < 2; ++j) acc[i][j] = v8f_zero();

  const int nIter = K >> 6;  // K-step 64

  auto stage = [&](int it, int buf) {
    const int k0 = it * 64;
    // A tile 64x64: 512 chunks of 8 bf16, 2 per thread
#pragma unroll
    for (int j = 0; j < 2; ++j) {
      int chk = tid * 2 + j;
      int row = chk >> 3, c = chk & 7;
      v8bf v = *reinterpret_cast<const v8bf*>(A + (size_t)(m0 + row) * K + k0 + c * 8);
      *reinterpret_cast<v8bf*>(&As[buf][row * 72 + c * 8]) = v;
    }
    // B tile 128x64: 1024 chunks, 4 per thread
#pragma unroll
    for (int j = 0; j < 4; ++j) {
      int chk = tid * 4 + j;
      int row = chk >> 3, c = chk & 7;
      v8bf v = *reinterpret_cast<const v8bf*>(B + (size_t)(n0 + row) * K + k0 + c * 8);
      *reinterpret_cast<v8bf*>(&Bs[buf][row * 72 + c * 8]) = v;
    }
  };

  stage(0, 0);

  for (int it = 0; it < nIter; ++it) {
    __syncthreads();
    if (it + 1 < nIter) {
      stage(it + 1, (it + 1) & 1);
      if (it + 2 < nIter) {  // prefetch tile it+2 into L2 (global_prefetch_b8)
        __builtin_prefetch(A + (size_t)(m0 + (tid >> 3)) * K + (it + 2) * 64, 0, 0);
        __builtin_prefetch(B + (size_t)(n0 + (tid >> 1)) * K + (it + 2) * 64, 0, 0);
      }
    }
    const int cur = it & 1;
#pragma unroll
    for (int ks = 0; ks < 2; ++ks) {
      v16bf af[2], bfm[2];
#pragma unroll
      for (int i = 0; i < 2; ++i) {
        int r = wm * 32 + i * 16 + nloc;
        af[i] = mkfrag(*reinterpret_cast<const v8bf*>(&As[cur][r * 72 + ks * 32 + half * 8]),
                       *reinterpret_cast<const v8bf*>(&As[cur][r * 72 + ks * 32 + 16 + half * 8]));
      }
#pragma unroll
      for (int j = 0; j < 2; ++j) {
        int r = wn * 32 + j * 16 + nloc;
        bfm[j] = mkfrag(*reinterpret_cast<const v8bf*>(&Bs[cur][r * 72 + ks * 32 + half * 8]),
                        *reinterpret_cast<const v8bf*>(&Bs[cur][r * 72 + ks * 32 + 16 + half * 8]));
      }
#pragma unroll
      for (int i = 0; i < 2; ++i)
#pragma unroll
        for (int j = 0; j < 2; ++j)
          acc[i][j] = WMMA_BF16(af[i], bfm[j], acc[i][j]);
    }
  }
}

// ---------------------------------------------------------------------------
// QKV GEMM: [8192 x 3072] = x_bf16[8192x1024] @ w_attn[3072x1024]^T
// Block-uniform routing: with 128-wide N tiles and C=1024, blockIdx.y>>3
// selects q (0), k (1) or v (2) for the WHOLE block -> no per-element branches.
// q,k stored [B,H,T,HD]; v stored transposed [B,H,HD,T] (bf16).
// ---------------------------------------------------------------------------
__global__ __launch_bounds__(256) void qkv_gemm_kernel(const __bf16* __restrict__ xb,
                                                       const __bf16* __restrict__ wab,
                                                       __bf16* __restrict__ qb,
                                                       __bf16* __restrict__ kb,
                                                       __bf16* __restrict__ vtb) {
  v8f acc[2][2];
  const int m0 = blockIdx.x * 64, n0 = blockIdx.y * 128;
  gemm_core(xb, wab, 1024, m0, n0, acc);

  const int tid = threadIdx.x, lane = tid & 31, nloc = lane & 15, half = lane >> 4;
  const int widx = tid >> 5, wm = widx >> 2, wn = widx & 3;
  const int which = blockIdx.y >> 3;      // uniform: 0=q, 1=k, 2=v
  const int cbase = (n0 & 1023) + wn * 32;

  if (which < 2) {
    __bf16* dst = (which == 0) ? qb : kb;
#pragma unroll
    for (int i = 0; i < 2; ++i)
#pragma unroll
      for (int j = 0; j < 2; ++j)
#pragma unroll
        for (int r = 0; r < 8; ++r) {
          int mg = m0 + wm * 32 + i * 16 + r + 8 * half;  // token in [0, B*T)
          int c  = cbase + j * 16 + nloc;                 // channel in [0, C)
          int bb = mg >> 11, t = mg & 2047, hh = c >> 6, d = c & 63;
          dst[((size_t)(bb * 16 + hh) * 2048 + t) * 64 + d] = (__bf16)acc[i][j][r];
        }
  } else {
#pragma unroll
    for (int i = 0; i < 2; ++i)
#pragma unroll
      for (int j = 0; j < 2; ++j)
#pragma unroll
        for (int r = 0; r < 8; ++r) {
          int mg = m0 + wm * 32 + i * 16 + r + 8 * half;
          int c  = cbase + j * 16 + nloc;
          int bb = mg >> 11, t = mg & 2047, hh = c >> 6, d = c & 63;
          vtb[((size_t)(bb * 16 + hh) * 64 + d) * 2048 + t] = (__bf16)acc[i][j][r];
        }
  }
}

// ---------------------------------------------------------------------------
// Flash-style causal attention: one wave per 32-row q block (two 16-row
// tiles), kv tiles of 32. K and V fragments are loaded once per kv tile and
// shared by both q tiles -> 16 WMMAs per kv iteration. Online softmax via
// shfl_xor row reductions; P bounced through per-wave LDS (C->A layout).
// ---------------------------------------------------------------------------
__global__ __launch_bounds__(256) void attn_kernel(const __bf16* __restrict__ qb,
                                                   const __bf16* __restrict__ kb,
                                                   const __bf16* __restrict__ vtb,
                                                   __bf16* __restrict__ yb) {
  __shared__ __bf16 Pbuf[8 * 2 * 16 * 40];  // 8 waves x 2 q-tiles x (16 x 40)

  const int tid = threadIdx.x, lane = tid & 31, nloc = lane & 15, half = lane >> 4;
  const int w = tid >> 5;
  const int wid = blockIdx.x * 8 + w;       // [0, 4096)
  const int qt = wid & 63, h = (wid >> 6) & 15, b = wid >> 10;
  const int q0 = qt * 32;

  const __bf16* Q  = qb  + ((size_t)(b * 16 + h) * 2048 + q0) * 64;
  const __bf16* Kp = kb  + (size_t)(b * 16 + h) * 2048 * 64;
  const __bf16* Vt = vtb + (size_t)(b * 16 + h) * 64 * 2048;
  __bf16* Pw = &Pbuf[w * 2 * 640];

  // Q fragments (A-layout), per q-tile: HD=64 = two K=32 fragments
  v16bf qf[2][2];
#pragma unroll
  for (int qq = 0; qq < 2; ++qq) {
    const __bf16* qr = Q + (qq * 16 + nloc) * 64;
    qf[qq][0] = mkfrag(*(const v8bf*)(qr + half * 8),      *(const v8bf*)(qr + 16 + half * 8));
    qf[qq][1] = mkfrag(*(const v8bf*)(qr + 32 + half * 8), *(const v8bf*)(qr + 48 + half * 8));
  }

  v8f acc[2][4];
  float run_m[2][8], run_l[2][8];
#pragma unroll
  for (int qq = 0; qq < 2; ++qq) {
#pragma unroll
    for (int nc = 0; nc < 4; ++nc) acc[qq][nc] = v8f_zero();
#pragma unroll
    for (int r = 0; r < 8; ++r) { run_m[qq][r] = -__builtin_inff(); run_l[qq][r] = 0.0f; }
  }

  const float scale = 0.125f;  // 1/sqrt(64)

  for (int kv0 = 0; kv0 <= q0 + 31; kv0 += 32) {
    // ---- K fragments, shared by both q tiles ----
    v16bf kf[2][2];
#pragma unroll
    for (int st = 0; st < 2; ++st) {
      const __bf16* kr = Kp + (size_t)(kv0 + st * 16 + nloc) * 64;
      kf[st][0] = mkfrag(*(const v8bf*)(kr + half * 8),      *(const v8bf*)(kr + 16 + half * 8));
      kf[st][1] = mkfrag(*(const v8bf*)(kr + 32 + half * 8), *(const v8bf*)(kr + 48 + half * 8));
    }

    // ---- S = Q K^T (8 WMMAs: 2 q-tiles x 2 kv-subtiles x 2 chained) ----
    v8f s[2][2];
#pragma unroll
    for (int qq = 0; qq < 2; ++qq)
#pragma unroll
      for (int st = 0; st < 2; ++st) {
        v8f z = v8f_zero();
        z = WMMA_BF16(qf[qq][0], kf[st][0], z);
        z = WMMA_BF16(qf[qq][1], kf[st][1], z);
        s[qq][st] = z;
      }

    // ---- causal mask + online softmax stats ----
    float alpha[2][8];
#pragma unroll
    for (int qq = 0; qq < 2; ++qq)
#pragma unroll
      for (int r = 0; r < 8; ++r) {
        int m  = r + 8 * half;
        int qg = q0 + qq * 16 + m;
        float sv0 = (kv0 + nloc      <= qg) ? s[qq][0][r] * scale : -__builtin_inff();
        float sv1 = (kv0 + 16 + nloc <= qg) ? s[qq][1][r] * scale : -__builtin_inff();
        float tm = fmaxf(sv0, sv1);
        tm = fmaxf(tm, __shfl_xor(tm, 1, 32));
        tm = fmaxf(tm, __shfl_xor(tm, 2, 32));
        tm = fmaxf(tm, __shfl_xor(tm, 4, 32));
        tm = fmaxf(tm, __shfl_xor(tm, 8, 32));
        float nm = fmaxf(run_m[qq][r], tm);
        float a  = __expf(run_m[qq][r] - nm);
        float p0 = __expf(sv0 - nm);
        float p1 = __expf(sv1 - nm);
        float rs = p0 + p1;
        rs += __shfl_xor(rs, 1, 32);
        rs += __shfl_xor(rs, 2, 32);
        rs += __shfl_xor(rs, 4, 32);
        rs += __shfl_xor(rs, 8, 32);
        run_l[qq][r] = run_l[qq][r] * a + rs;
        run_m[qq][r] = nm;
        alpha[qq][r] = a;
        Pw[qq * 640 + m * 40 + nloc]      = (__bf16)p0;
        Pw[qq * 640 + m * 40 + 16 + nloc] = (__bf16)p1;
      }

    // rescale running output
#pragma unroll
    for (int qq = 0; qq < 2; ++qq)
#pragma unroll
      for (int nc = 0; nc < 4; ++nc)
#pragma unroll
        for (int r = 0; r < 8; ++r) acc[qq][nc][r] *= alpha[qq][r];

    // LDS writes -> reads within the wave (DS is in-order; keep compiler honest)
    asm volatile("s_wait_dscnt 0x0" ::: "memory");

    // ---- O += P @ V (V fragments shared by both q tiles: 8 WMMAs) ----
    v16bf pf[2];
#pragma unroll
    for (int qq = 0; qq < 2; ++qq)
      pf[qq] = mkfrag(*(const v8bf*)(&Pw[qq * 640 + nloc * 40 + half * 8]),
                      *(const v8bf*)(&Pw[qq * 640 + nloc * 40 + 16 + half * 8]));
#pragma unroll
    for (int nc = 0; nc < 4; ++nc) {
      const __bf16* vr = Vt + (size_t)(nc * 16 + nloc) * 2048 + kv0;
      v16bf vf = mkfrag(*(const v8bf*)(vr + half * 8), *(const v8bf*)(vr + 16 + half * 8));
#pragma unroll
      for (int qq = 0; qq < 2; ++qq)
        acc[qq][nc] = WMMA_BF16(pf[qq], vf, acc[qq][nc]);
    }
  }

  // ---- normalize and write y in [B,T,C] layout (bf16 for the proj GEMM) ----
#pragma unroll
  for (int qq = 0; qq < 2; ++qq)
#pragma unroll
    for (int r = 0; r < 8; ++r) {
      float inv = 1.0f / run_l[qq][r];
      int t = q0 + qq * 16 + r + 8 * half;
      size_t base = ((size_t)b * 2048 + t) * 1024 + h * 64;
#pragma unroll
      for (int nc = 0; nc < 4; ++nc)
        yb[base + nc * 16 + nloc] = (__bf16)(acc[qq][nc][r] * inv);
    }
}

// ---------------------------------------------------------------------------
// Output projection: out[8192x1024] (fp32) = y_bf16 @ w_proj[1024x1024]^T
// ---------------------------------------------------------------------------
__global__ __launch_bounds__(256) void proj_gemm_kernel(const __bf16* __restrict__ yb,
                                                        const __bf16* __restrict__ wpb,
                                                        float* __restrict__ out) {
  v8f acc[2][2];
  const int m0 = blockIdx.x * 64, n0 = blockIdx.y * 128;
  gemm_core(yb, wpb, 1024, m0, n0, acc);

  const int tid = threadIdx.x, lane = tid & 31, nloc = lane & 15, half = lane >> 4;
  const int widx = tid >> 5, wm = widx >> 2, wn = widx & 3;
#pragma unroll
  for (int i = 0; i < 2; ++i)
#pragma unroll
    for (int j = 0; j < 2; ++j)
#pragma unroll
      for (int r = 0; r < 8; ++r) {
        int mg = m0 + wm * 32 + i * 16 + r + 8 * half;
        int ng = n0 + wn * 32 + j * 16 + nloc;
        out[(size_t)mg * 1024 + ng] = acc[i][j][r];
      }
}

// ---------------------------------------------------------------------------
// Launch: convert -> QKV GEMM -> attention -> proj GEMM (all on `stream`).
// Workspace layout (bytes), total ~88 MB:
//   xb 0..16M | wab ..22M | wpb ..24M | qb ..40M | kb ..56M | vtb ..72M | yb ..88M
// ---------------------------------------------------------------------------
extern "C" void kernel_launch(void* const* d_in, const int* in_sizes, int n_in,
                              void* d_out, int out_size, void* d_ws, size_t ws_size,
                              hipStream_t stream) {
  (void)in_sizes; (void)n_in; (void)out_size; (void)ws_size;

  const float* x      = (const float*)d_in[0];  // [4,2048,1024]
  const float* w_attn = (const float*)d_in[1];  // [3072,1024]
  const float* w_proj = (const float*)d_in[2];  // [1024,1024]
  float* out = (float*)d_out;                   // [4,2048,1024]

  char* ws = (char*)d_ws;
  __bf16* xb  = (__bf16*)(ws);
  __bf16* wab = (__bf16*)(ws + 16777216);
  __bf16* wpb = (__bf16*)(ws + 23068672);
  __bf16* qb  = (__bf16*)(ws + 25165824);
  __bf16* kb  = (__bf16*)(ws + 41943040);
  __bf16* vtb = (__bf16*)(ws + 58720256);
  __bf16* yb  = (__bf16*)(ws + 75497472);

  cvt_f32_bf16_kernel<<<8388608 / 1024, 256, 0, stream>>>(x, xb);
  cvt_f32_bf16_kernel<<<3145728 / 1024, 256, 0, stream>>>(w_attn, wab);
  cvt_f32_bf16_kernel<<<1048576 / 1024, 256, 0, stream>>>(w_proj, wpb);

  qkv_gemm_kernel<<<dim3(128, 24), 256, 0, stream>>>(xb, wab, qb, kb, vtb);
  attn_kernel<<<512, 256, 0, stream>>>(qb, kb, vtb, yb);
  proj_gemm_kernel<<<dim3(128, 8), 256, 0, stream>>>(yb, wpb, out);
}
